// BatchedQNodeLayer_16561393893459
// MI455X (gfx1250) — compile-verified, matrix operand order checked by reading, not run on hardware
//
#include <hip/hip_runtime.h>

// ---------------------------------------------------------------------------
// MI455X (gfx1250): batched 8-qubit QNode as one real GEMM on WMMA.
//
// ev[b] = sum_i sign(i) * | (U(theta) psi(x_b))_i |^2
//   psi(x_b): product state, psi[k] = m(k) * (-i)^popcount(k)   (cheap VALU)
//   U(theta): fixed 256x256 complex unitary (batch independent, kernel 1)
//   complex matvec -> real GEMM A[B x 512] * W[512 x 512] (re/im expansion),
//   Z0 sign is uniform per 16-wide N tile and folded into the epilogue.
//
// 68.7 GFLOP, ~5 MB HBM -> matrix-core bound -> v_wmma_f32_16x16x32_f16.
// Per wave: 32 batch rows (two M-tiles sharing every B fragment -> halves L2
// traffic for W). All 32 A fragments (256 VGPRs) live in registers for the
// whole kernel; all 16 B fragments of an N-tile are loaded up-front into a
// register array (32 loads in flight, in-order returns -> progressive
// s_wait_loadcnt), so load latency is paid once per 32 WMMAs, not per 2.
// ---------------------------------------------------------------------------

typedef _Float16 v16h __attribute__((ext_vector_type(16)));
typedef _Float16 v8h  __attribute__((ext_vector_type(8)));
typedef float    v8f  __attribute__((ext_vector_type(8)));

#define NQ    8
#define DIM   256      // 2^8
#define KDIM  512      // real-expanded K
#define APAD  8        // LDS row padding (halves) -> bank-conflict-free
#define NT_TILES 32    // 512 / 16
#define KT_TILES 16    // 512 / 32

union V16 { v16h v; v8h h[2]; };

// Fragment index map for the 16-bit 32x16 WMMA B operand (wave32):
// lanes 0-15 carry K = {0..7, 16..23}, lanes 16-31 carry K = {8..15, 24..31}
// (mirror of the documented A layout), N = lane%16. Stored so each lane's
// 16 halves are contiguous -> two 16B global loads per fragment.
__device__ __forceinline__ int wfrag_index(int k, int n) {
    int kt = k >> 5, kk = k & 31;
    int nt = n >> 4, nn = n & 15;
    int lane = nn + (((kk >> 3) & 1) << 4);
    int v    = ((kk >> 4) << 2) + ((kk & 7) >> 1);
    int t    = kk & 1;
    return (((kt * 32 + nt) * 32) + lane) * 16 + (v * 2 + t);
}

// ---------------------------------------------------------------------------
// Kernel 1: build U(theta) column-by-column (256 blocks, one column each,
// simulated in LDS), scatter into pre-swizzled f16 W fragments in d_ws.
// Fully rewrites the workspace every launch (deterministic).
// ---------------------------------------------------------------------------
__global__ void build_wfrag(const float* __restrict__ theta,
                            _Float16* __restrict__ wf) {
    __shared__ float ar[DIM];
    __shared__ float ai[DIM];
    const int t = threadIdx.x;       // 64 threads
    const int j = blockIdx.x;        // column / initial basis state

    for (int i = t; i < DIM; i += 64) { ar[i] = (i == j) ? 1.f : 0.f; ai[i] = 0.f; }
    __syncthreads();

    for (int l = 0; l < 2; ++l) {
        // Rot(phi, th, om) = RZ(om) RY(th) RZ(phi) on each wire
        for (int w = 0; w < NQ; ++w) {
            const float phi = theta[(l * NQ + w) * 3 + 0];
            const float th  = theta[(l * NQ + w) * 3 + 1];
            const float om  = theta[(l * NQ + w) * 3 + 2];
            const float cc = cosf(0.5f * th), ss = sinf(0.5f * th);
            const float ca = cosf(0.5f * (phi + om)), sa = sinf(0.5f * (phi + om));
            const float cb = cosf(0.5f * (phi - om)), sb = sinf(0.5f * (phi - om));
            const float g00r =  ca * cc, g00i = -sa * cc;
            const float g01r = -cb * ss, g01i = -sb * ss;
            const float g10r =  cb * ss, g10i = -sb * ss;
            const float g11r =  ca * cc, g11i =  sa * cc;
            const int pw = 7 - w;      // wire 0 = MSB of flat index
            for (int p = t; p < 128; p += 64) {
                const int lo = p & ((1 << pw) - 1);
                const int i0 = lo | ((p >> pw) << (pw + 1));
                const int i1 = i0 | (1 << pw);
                const float a0r = ar[i0], a0i = ai[i0];
                const float a1r = ar[i1], a1i = ai[i1];
                ar[i0] = g00r * a0r - g00i * a0i + g01r * a1r - g01i * a1i;
                ai[i0] = g00r * a0i + g00i * a0r + g01r * a1i + g01i * a1r;
                ar[i1] = g10r * a0r - g10i * a0i + g11r * a1r - g11i * a1i;
                ai[i1] = g10r * a0i + g10i * a0r + g11r * a1i + g11i * a1r;
            }
            __syncthreads();
        }
        const int r = (l % 7) + 1;     // CNOT ring range
        for (int w = 0; w < NQ; ++w) {
            const int pc = 7 - w;
            const int pt = 7 - ((w + r) & 7);
            int i = 0, tt = t;                 // spread 6 free bits
            for (int b = 0; b < 8; ++b) {
                if (b == pc || b == pt) continue;
                i |= (tt & 1) << b; tt >>= 1;
            }
            i |= (1 << pc);                    // ctrl = 1, tgt = 0
            const int i2 = i | (1 << pt);
            const float r0 = ar[i], q0 = ai[i];
            const float r1 = ar[i2], q1 = ai[i2];
            ar[i] = r1; ai[i] = q1; ar[i2] = r0; ai[i2] = q0;
            __syncthreads();
        }
    }

    for (int i = t; i < DIM; i += 64) {        // amp[i] = U[i, j]
        const float Ur = ar[i], Ui = ai[i];
        wf[wfrag_index(j,       2 * i    )] = (_Float16)  Ur;
        wf[wfrag_index(j,       2 * i + 1)] = (_Float16)  Ui;
        wf[wfrag_index(j + DIM, 2 * i    )] = (_Float16)(-Ui);
        wf[wfrag_index(j + DIM, 2 * i + 1)] = (_Float16)  Ur;
    }
}

// ---------------------------------------------------------------------------
// Kernel 2: one wave per block, 32 batch rows per wave (two WMMA M-tiles).
// ---------------------------------------------------------------------------
__global__ void __launch_bounds__(32)
qgemm_wmma(const float* __restrict__ x,
           const _Float16* __restrict__ wf,
           float* __restrict__ out) {
    __shared__ _Float16 As[32][KDIM + APAD];   // padded rows: no bank conflicts
    const int lane = threadIdx.x & 31;
    const int b0   = blockIdx.x * 32;

    // ---- build psi: one batch row per lane, 256 complex amplitudes ----
    {
        const float* xp = x + (size_t)(b0 + lane) * NQ;
        float fc[8], fs[8];
        #pragma unroll
        for (int w = 0; w < NQ; ++w) {        // bit position b holds wire 7-b
            const float a = 0.5f * xp[7 - w];
            fc[w] = cosf(a); fs[w] = sinf(a);
        }
        for (int k = 0; k < DIM; ++k) {
            float m = 1.f; int pc = 0;
            #pragma unroll
            for (int b = 0; b < 8; ++b) {
                if ((k >> b) & 1) { m *= fs[b]; ++pc; } else { m *= fc[b]; }
            }
            pc &= 3;                          // psi = m * (-i)^pc
            const float pr = (pc == 0) ? 1.f : (pc == 2 ? -1.f : 0.f);
            const float pi = (pc == 1) ? -1.f : (pc == 3 ? 1.f : 0.f);
            As[lane][k]       = (_Float16)(m * pr);
            As[lane][DIM + k] = (_Float16)(m * pi);
        }
    }
    __syncthreads();

    // ---- hoist ALL A fragments into registers (2 M-tiles x 16 K-tiles) ----
    const int arow = lane & 15;
    const int alo  = (lane < 16) ? 0 : 8;     // documented A-matrix K split
    v16h A0[KT_TILES], A1[KT_TILES];
    #pragma unroll
    for (int kt = 0; kt < KT_TILES; ++kt) {
        const _Float16* ap0 = &As[arow][kt * 32 + alo];
        const _Float16* ap1 = &As[arow + 16][kt * 32 + alo];
        V16 u0, u1;
        u0.h[0] = *reinterpret_cast<const v8h*>(ap0);
        u0.h[1] = *reinterpret_cast<const v8h*>(ap0 + 16);
        u1.h[0] = *reinterpret_cast<const v8h*>(ap1);
        u1.h[1] = *reinterpret_cast<const v8h*>(ap1 + 16);
        A0[kt] = u0.v;
        A1[kt] = u1.v;
    }

    float p0[8], p1[8];
    #pragma unroll
    for (int v = 0; v < 8; ++v) { p0[v] = 0.f; p1[v] = 0.f; }

    #pragma unroll 1
    for (int nt = 0; nt < NT_TILES; ++nt) {
        if (nt + 1 < NT_TILES)                // pull next N-tile's W into cache
            __builtin_prefetch(wf + (((size_t)(nt + 1) * 32) + lane) * 16, 0, 1);

        // Issue ALL 32 B loads of this N-tile first (independent, in-order
        // returns) so the WMMA stream below waits with slack, not to zero.
        V16 Bf[KT_TILES];
        #pragma unroll
        for (int kt = 0; kt < KT_TILES; ++kt) {
            const _Float16* bp = wf + (((size_t)(kt * 32 + nt) * 32) + lane) * 16;
            Bf[kt].h[0] = *reinterpret_cast<const v8h*>(bp);
            Bf[kt].h[1] = *reinterpret_cast<const v8h*>(bp + 8);
        }
        v8f acc0 = {}, acc1 = {};
        #pragma unroll
        for (int kt = 0; kt < KT_TILES; ++kt) {
            acc0 = __builtin_amdgcn_wmma_f32_16x16x32_f16(
                       false, A0[kt], false, Bf[kt].v, (short)0, acc0, false, false);
            acc1 = __builtin_amdgcn_wmma_f32_16x16x32_f16(
                       false, A1[kt], false, Bf[kt].v, (short)0, acc1, false, false);
        }
        // PauliZ(0): output complex index i = nt*8 + n/2; i>=128 <=> nt>=16
        const float sgn = (nt < 16) ? 1.f : -1.f;
        #pragma unroll
        for (int v = 0; v < 8; ++v) {
            p0[v] += sgn * acc0[v] * acc0[v];
            p1[v] += sgn * acc1[v] * acc1[v];
        }
    }

    // ---- reduce over the 16 columns held by each 16-lane half (wave32) ----
    #pragma unroll
    for (int m = 1; m <= 8; m <<= 1) {
        #pragma unroll
        for (int v = 0; v < 8; ++v) {
            p0[v] += __shfl_xor(p0[v], m, 32);
            p1[v] += __shfl_xor(p1[v], m, 32);
        }
    }
    // D layout: vgpr v, lanes 0-15 -> row v; lanes 16-31 -> row v+8
    if ((lane & 15) < 8) {
        const int rr = (lane & 15) + ((lane >> 4) << 3);
        out[b0 + rr]      = 0.5f * (p0[lane & 15] + 1.f);
        out[b0 + 16 + rr] = 0.5f * (p1[lane & 15] + 1.f);
    }
}

// ---------------------------------------------------------------------------
extern "C" void kernel_launch(void* const* d_in, const int* in_sizes, int n_in,
                              void* d_out, int out_size, void* d_ws, size_t ws_size,
                              hipStream_t stream) {
    const float* x     = (const float*)d_in[0];   // [B, 8] f32
    const float* theta = (const float*)d_in[1];   // [2, 8, 3] f32
    _Float16*    wf    = (_Float16*)d_ws;         // 512x512 f16 fragments (512 KB)
    float*       out   = (float*)d_out;           // [B, 1] f32

    const int batch = in_sizes[0] / NQ;           // 131072

    build_wfrag<<<DIM, 64, 0, stream>>>(theta, wf);
    qgemm_wmma<<<batch / 32, 32, 0, stream>>>(x, wf, out);
}
